// Transformer_82360292868229
// MI455X (gfx1250) — compile-verified, matrix operand order checked by reading, not run on hardware
//
#include <hip/hip_runtime.h>
#include <math.h>

#define D_   1024
#define H_   16
#define HD_  64
#define SCALE_ 0.125f           // 1/sqrt(64)
#define NEG_ (-3.402823466e38f) // finfo(float32).min, as in the reference

typedef __attribute__((ext_vector_type(16))) __bf16 v16bf;
typedef __attribute__((ext_vector_type(8)))  float  v8f;

union FragB { v16bf v; uint4 q[2]; unsigned short us[16]; };
union AccF  { v8f v; float f[8]; };

// hardware bf16 convert (fptrunc -> v_cvt bf16 on gfx1250)
__device__ __forceinline__ unsigned short f2bf(float f) {
  union { __bf16 h; unsigned short u; } c;
  c.h = (__bf16)f;
  return c.u;
}

// ---------------------------------------------------------------------------
// One-time preps (memory-bound, trivially small vs the GEMMs):
//   conv_bf16  : activations f32 -> bf16, row-major
//   prep_weight: W (KxN, f32) -> Wt (NxK, bf16)  via LDS tile transpose
// ---------------------------------------------------------------------------
__global__ __launch_bounds__(256) void conv_bf16(const float* __restrict__ X,
                                                 unsigned short* __restrict__ Y,
                                                 int n4) {
  int i = blockIdx.x * 256 + threadIdx.x;
  if (i < n4) {
    float4 v = ((const float4*)X)[i];
    union { unsigned short u[4]; uint2 q; } o;
    o.u[0] = f2bf(v.x); o.u[1] = f2bf(v.y);
    o.u[2] = f2bf(v.z); o.u[3] = f2bf(v.w);
    ((uint2*)Y)[i] = o.q;
  }
}

__global__ __launch_bounds__(256) void prep_weight(const float* __restrict__ W,
                                                   unsigned short* __restrict__ Wt) {
  __shared__ float t[32][33];
  const int tx = threadIdx.x & 31, ty = threadIdx.x >> 5; // 32 x 8
  const int k0 = blockIdx.y * 32, n0 = blockIdx.x * 32;
#pragma unroll
  for (int j = 0; j < 4; ++j)
    t[ty + j * 8][tx] = W[(size_t)(k0 + ty + j * 8) * D_ + n0 + tx];
  __syncthreads();
#pragma unroll
  for (int j = 0; j < 4; ++j)
    Wt[(size_t)(n0 + ty + j * 8) * D_ + k0 + tx] = f2bf(t[tx][ty + j * 8]);
}

// ---------------------------------------------------------------------------
// Register-resident WMMA GEMM: Y = A(bf16, Mx1024) * Wt^T + bias.
// A is row-major bf16; Wt is the weight pre-transposed (NxK bf16), so both
// fragment operands are contiguous global_load_b128.  Each wave owns a
// 16x64 output tile (1 A-frag reused by 4 WMMAs per 32-wide k-step); a
// 256-thread block is 8 independent waves covering 128 rows x 64 cols.
// MODE 0: bf16 dst[((b*H+h)*S+s)*HD+hd]   (Q/K head layout)
// MODE 2: bf16 dst[((b*H+h)*HD+hd)*S+s]   (V transposed head layout)
// MODE 3: f32  dst[row*D + col]           (final output + bias)
// ---------------------------------------------------------------------------
template <int MODE>
__global__ __launch_bounds__(256) void wmma_gemm(const unsigned short* __restrict__ A,
                                                 const unsigned short* __restrict__ Wt,
                                                 const float* __restrict__ bias,
                                                 void* __restrict__ dstv,
                                                 int S) {
  const int lane = threadIdx.x & 31;
  const int wv   = threadIdx.x >> 5;
  const int half = lane >> 4;
  const int l16  = lane & 15;
  const int row0 = (blockIdx.y * 8 + wv) * 16;
  const int col0 = blockIdx.x * 64;

  AccF acc[4];
#pragma unroll
  for (int t = 0; t < 4; ++t)
#pragma unroll
    for (int r = 0; r < 8; ++r) acc[t].f[r] = 0.0f;

  const unsigned short* ap0 = A  + (size_t)(row0 + l16) * D_ + half * 8;
  const unsigned short* bp0 = Wt + (size_t)(col0 + l16) * D_ + half * 8;

  for (int k0 = 0; k0 < D_; k0 += 32) {
    FragB a;
    a.q[0] = *(const uint4*)(ap0 + k0);
    a.q[1] = *(const uint4*)(ap0 + k0 + 16);
    if (k0 + 32 < D_)
      __builtin_prefetch(ap0 + k0 + 32, 0, 3);
#pragma unroll
    for (int nt = 0; nt < 4; ++nt) {
      FragB b;
      const unsigned short* bp = bp0 + (size_t)nt * 16 * D_ + k0;
      b.q[0] = *(const uint4*)(bp);
      b.q[1] = *(const uint4*)(bp + 16);
      acc[nt].v = __builtin_amdgcn_wmma_f32_16x16x32_bf16(
          false, a.v, false, b.v, (short)0, acc[nt].v, false, false);
    }
  }

#pragma unroll
  for (int nt = 0; nt < 4; ++nt) {
    const int col = col0 + nt * 16 + l16;
    const float bv = bias[col];
    const int h = col >> 6, hd = col & 63;
#pragma unroll
    for (int r = 0; r < 8; ++r) {
      int row = row0 + r + 8 * half;
      float y = acc[nt].f[r] + bv;
      if (MODE == 3) {
        ((float*)dstv)[(size_t)row * D_ + col] = y;
      } else {
        int bb = row / S;
        int s  = row - bb * S;
        unsigned short* dst = (unsigned short*)dstv;
        if (MODE == 0)
          dst[(((size_t)bb * H_ + h) * S + s) * HD_ + hd] = f2bf(y);
        else
          dst[(((size_t)bb * H_ + h) * HD_ + hd) * S + s] = f2bf(y);
      }
    }
  }
}

// ---------------------------------------------------------------------------
// Flash attention: block = 128 threads = 4 independent waves; each wave owns
// 16 query rows, streams keys in chunks of 32 with online softmax.  Q frags
// and the 16x64 f32 accumulator stay in registers; P round-trips through a
// padded per-wave LDS tile to convert C-layout -> A-layout (same-wave DS ops
// are in-order).  Row statistics use shfl_xor within the 16-lane halves,
// matching the WMMA C layout (row r in VGPR r across lanes).
// ---------------------------------------------------------------------------
__global__ __launch_bounds__(128) void attn_kernel(const unsigned short* __restrict__ Qh,
                                                   const unsigned short* __restrict__ Kh,
                                                   const unsigned short* __restrict__ Vt,
                                                   const unsigned char* __restrict__ mask,
                                                   unsigned short* __restrict__ Ctx,
                                                   int S) {
  __shared__ __align__(16) unsigned short Plds[4 * 16 * 40];
  const int lane = threadIdx.x & 31;
  const int w    = threadIdx.x >> 5;
  const int half = lane >> 4;
  const int l16  = lane & 15;
  const int bh   = blockIdx.y;
  const int b    = bh / H_;
  const int h    = bh - b * H_;
  const int q0   = blockIdx.x * 64 + w * 16;

  const unsigned short* Qbase = Qh + (size_t)bh * S * HD_;
  const unsigned short* Kbase = Kh + (size_t)bh * S * HD_;
  const unsigned short* Vbase = Vt + (size_t)bh * HD_ * S;

  FragB aq[2];
#pragma unroll
  for (int ks = 0; ks < 2; ++ks) {
    const unsigned short* p = Qbase + (size_t)(q0 + l16) * HD_ + ks * 32 + half * 8;
    aq[ks].q[0] = *(const uint4*)(p);
    aq[ks].q[1] = *(const uint4*)(p + 16);
  }

  AccF acc[4];
#pragma unroll
  for (int t = 0; t < 4; ++t)
#pragma unroll
    for (int r = 0; r < 8; ++r) acc[t].f[r] = 0.0f;

  float m_i[8], l_i[8];
#pragma unroll
  for (int r = 0; r < 8; ++r) { m_i[r] = NEG_; l_i[r] = 0.0f; }

  unsigned short* Pw = &Plds[w * 16 * 40];

  for (int c = 0; c < S; c += 32) {
    AccF sc[2];
#pragma unroll
    for (int nt = 0; nt < 2; ++nt) {
#pragma unroll
      for (int r = 0; r < 8; ++r) sc[nt].f[r] = 0.0f;
#pragma unroll
      for (int ks = 0; ks < 2; ++ks) {
        FragB bk;
        const unsigned short* p =
            Kbase + (size_t)(c + nt * 16 + l16) * HD_ + ks * 32 + half * 8;
        bk.q[0] = *(const uint4*)(p);
        bk.q[1] = *(const uint4*)(p + 16);
        sc[nt].v = __builtin_amdgcn_wmma_f32_16x16x32_bf16(
            false, aq[ks].v, false, bk.v, (short)0, sc[nt].v, false, false);
      }
    }
#pragma unroll
    for (int nt = 0; nt < 2; ++nt) {
#pragma unroll
      for (int r = 0; r < 8; ++r) {
        int qr = q0 + r + 8 * half;
        int kc = c + nt * 16 + l16;
        float v = sc[nt].f[r] * SCALE_;
        sc[nt].f[r] = mask[(size_t)qr * S + kc] ? v : NEG_;
      }
    }
#pragma unroll
    for (int r = 0; r < 8; ++r) {
      float cm = fmaxf(sc[0].f[r], sc[1].f[r]);
      cm = fmaxf(cm, __shfl_xor(cm, 1, 32));
      cm = fmaxf(cm, __shfl_xor(cm, 2, 32));
      cm = fmaxf(cm, __shfl_xor(cm, 4, 32));
      cm = fmaxf(cm, __shfl_xor(cm, 8, 32));
      float mn = fmaxf(m_i[r], cm);
      float so = __expf(m_i[r] - mn);
      float p0 = __expf(sc[0].f[r] - mn);
      float p1 = __expf(sc[1].f[r] - mn);
      float rs = p0 + p1;
      rs += __shfl_xor(rs, 1, 32);
      rs += __shfl_xor(rs, 2, 32);
      rs += __shfl_xor(rs, 4, 32);
      rs += __shfl_xor(rs, 8, 32);
      l_i[r] = l_i[r] * so + rs;
      m_i[r] = mn;
#pragma unroll
      for (int t = 0; t < 4; ++t) acc[t].f[r] *= so;
      int prow = r + 8 * half;
      Pw[prow * 40 + l16]      = f2bf(p0);
      Pw[prow * 40 + 16 + l16] = f2bf(p1);
    }
    FragB pa;
    {
      const unsigned short* p = &Pw[l16 * 40 + half * 8];
      pa.q[0] = *(const uint4*)(p);
      pa.q[1] = *(const uint4*)(p + 16);
    }
#pragma unroll
    for (int nt = 0; nt < 4; ++nt) {
      FragB bv;
      const unsigned short* p =
          Vbase + (size_t)(nt * 16 + l16) * S + c + half * 8;
      bv.q[0] = *(const uint4*)(p);
      bv.q[1] = *(const uint4*)(p + 16);
      acc[nt].v = __builtin_amdgcn_wmma_f32_16x16x32_bf16(
          false, pa.v, false, bv.v, (short)0, acc[nt].v, false, false);
    }
  }

#pragma unroll
  for (int r = 0; r < 8; ++r) {
    float inv = (l_i[r] > 0.0f) ? (1.0f / l_i[r]) : 0.0f;
    int s = q0 + r + 8 * half;
#pragma unroll
    for (int nt = 0; nt < 4; ++nt) {
      int d = h * HD_ + nt * 16 + l16;
      Ctx[((size_t)b * S + s) * D_ + d] = f2bf(acc[nt].f[r] * inv);
    }
  }
}

// ---------------------------------------------------------------------------
extern "C" void kernel_launch(void* const* d_in, const int* in_sizes, int n_in,
                              void* d_out, int out_size, void* d_ws, size_t ws_size,
                              hipStream_t stream) {
  const float* Xq = (const float*)d_in[0];
  const float* Xk = (const float*)d_in[1];
  const float* Xv = (const float*)d_in[2];
  const unsigned char* mask = (const unsigned char*)d_in[3];
  const float* Wq = (const float*)d_in[4];
  const float* bq = (const float*)d_in[5];
  const float* Wk = (const float*)d_in[6];
  const float* bk = (const float*)d_in[7];
  const float* Wv = (const float*)d_in[8];
  const float* bv = (const float*)d_in[9];
  const float* Wo = (const float*)d_in[10];
  const float* bo = (const float*)d_in[11];

  int S = (int)(sqrt((double)in_sizes[3]) + 0.5); // mask is (S,S)
  int B = in_sizes[0] / (S * D_);
  int BS = B * S;

  // workspace layout (Abuf recycled: Xq/Xk/Xv bf16 input, then ctx)
  unsigned short* Abuf = (unsigned short*)d_ws;        // BS*D bf16
  unsigned short* Wqt  = Abuf + (size_t)BS * D_;       // D*D bf16 each
  unsigned short* Wkt  = Wqt + (size_t)D_ * D_;
  unsigned short* Wvt  = Wkt + (size_t)D_ * D_;
  unsigned short* Wot  = Wvt + (size_t)D_ * D_;
  unsigned short* Qh   = Wot + (size_t)D_ * D_;        // (B,H,S,HD) bf16
  unsigned short* Kh   = Qh + (size_t)BS * D_;         // (B,H,S,HD) bf16
  unsigned short* Vt   = Kh + (size_t)BS * D_;         // (B,H,HD,S) bf16
  (void)ws_size; (void)n_in; (void)out_size;

  dim3 pb(256), pg(D_ / 32, D_ / 32);
  prep_weight<<<pg, pb, 0, stream>>>(Wq, Wqt);
  prep_weight<<<pg, pb, 0, stream>>>(Wk, Wkt);
  prep_weight<<<pg, pb, 0, stream>>>(Wv, Wvt);
  prep_weight<<<pg, pb, 0, stream>>>(Wo, Wot);

  int n4 = BS * D_ / 4;
  dim3 cg((n4 + 255) / 256), cb(256);
  dim3 gb(256), gg(D_ / 64, BS / 128);

  conv_bf16<<<cg, cb, 0, stream>>>(Xq, Abuf, n4);
  wmma_gemm<0><<<gg, gb, 0, stream>>>(Abuf, Wqt, bq, Qh, S);
  conv_bf16<<<cg, cb, 0, stream>>>(Xk, Abuf, n4);
  wmma_gemm<0><<<gg, gb, 0, stream>>>(Abuf, Wkt, bk, Kh, S);
  conv_bf16<<<cg, cb, 0, stream>>>(Xv, Abuf, n4);
  wmma_gemm<2><<<gg, gb, 0, stream>>>(Abuf, Wvt, bv, Vt, S);

  dim3 ab(128), ag(S / 64, (unsigned)B * H_);
  attn_kernel<<<ag, ab, 0, stream>>>(Qh, Kh, Vt, mask, Abuf /*ctx*/, S);

  wmma_gemm<3><<<gg, gb, 0, stream>>>(Abuf, Wot, bo, d_out, S);
}
